// Model_28329604285035
// MI455X (gfx1250) — matrix operand order repaired and checked by reading, not compiled, not run on hardware
//
#include <hip/hip_runtime.h>

typedef __bf16 v16bf __attribute__((ext_vector_type(16)));
typedef __bf16 v8bf  __attribute__((ext_vector_type(8)));
typedef __bf16 v4bf  __attribute__((ext_vector_type(4)));
typedef float  v8f   __attribute__((ext_vector_type(8)));

#define N_MI 100000
#define N_DI 50000
#define E_EDGES 500000
#define L_LABELS 200000

union frag16 { v16bf v; v8bf h[2]; };

// ---------------------------------------------------------------------------
// Utility kernels
// ---------------------------------------------------------------------------
__global__ void zero_kernel(float* __restrict__ p, long long n) {
    long long i = (long long)blockIdx.x * blockDim.x + threadIdx.x;
    if (i < n) p[i] = 0.0f;
}

// fp32 -> bf16 streaming conversion, 4 elements/thread (float4 in, 8B out)
__global__ void cvt_bf16_kernel(const float* __restrict__ in,
                                __bf16* __restrict__ out, long long n4) {
    long long i = (long long)blockIdx.x * blockDim.x + threadIdx.x;
    if (i < n4) {
        const float4 v = ((const float4*)in)[i];
        v4bf o;
        o[0] = (__bf16)v.x; o[1] = (__bf16)v.y;
        o[2] = (__bf16)v.z; o[3] = (__bf16)v.w;
        ((v4bf*)out)[i] = o;
    }
}

__global__ void degree_kernel(const int* __restrict__ src, const int* __restrict__ dst,
                              float* deg_mi, float* deg_di, int E) {
    int i = blockIdx.x * blockDim.x + threadIdx.x;
    if (i < E) {
        atomicAdd(&deg_mi[src[i]], 1.0f);
        atomicAdd(&deg_di[dst[i]], 1.0f);
    }
}

// thread f owns feature lane f; block streams over edges
__global__ void scatter_kernel(const float* __restrict__ feat,
                               const int* __restrict__ sidx,
                               const int* __restrict__ didx,
                               float* agg, int E, int F) {
    int f = threadIdx.x;  // blockDim.x == F
    for (int e = blockIdx.x; e < E; e += gridDim.x) {
        int s = sidx[e];
        int d = didx[e];
        atomicAdd(&agg[(size_t)d * F + f], feat[(size_t)s * F + f]);
    }
}

__global__ void mean_kernel(float* agg, const float* __restrict__ deg,
                            long long total, int F) {
    long long i = (long long)blockIdx.x * blockDim.x + threadIdx.x;
    if (i < total) {
        int n = (int)(i / F);
        agg[i] = agg[i] / fmaxf(deg[n], 1.0f);
    }
}

// ---------------------------------------------------------------------------
// GEMM: C[M,N] = A[M,K] @ W[N,K]^T (+bias[N]) (+addend[M,N]) (+ReLU)
// A, W are bf16 row-major in global; fragments loaded directly (no LDS).
// Block = 128 threads = 4 waves; block tile 128(M) x 64(N);
// each wave: 32x64 = 2x4 WMMA tiles, K stepped by 32.
// Out-of-range M rows are CLAMPED on load (always in-bounds, no divergence)
// and discarded by the store guard in the epilogue.
// Requires K % 32 == 0, N % 64 == 0.
// ---------------------------------------------------------------------------
__launch_bounds__(128)
__global__ void gemm_wmma_bf16(const __bf16* __restrict__ A,
                               const __bf16* __restrict__ W,
                               const float* __restrict__ bias,
                               const float* addend,
                               float* C,
                               int M, int N, int K, int relu) {
    const int m0   = blockIdx.x * 128;
    const int n0   = blockIdx.y * 64;
    const int lane = threadIdx.x & 31;
    const int wave = threadIdx.x >> 5;
    const int half = lane >> 4;
    const int l16  = lane & 15;

    const int rA0 = m0 + wave * 32 + l16;  // A-frag row, lower 16x32 tile
    const int rA1 = rA0 + 16;              // upper tile
    const int rA0c = (rA0 < M) ? rA0 : (M - 1);   // clamp: loads always valid
    const int rA1c = (rA1 < M) ? rA1 : (M - 1);

    v8f acc[2][4] = {};

    const __bf16* pa0 = A + (size_t)rA0c * K + half * 8;
    const __bf16* pa1 = A + (size_t)rA1c * K + half * 8;

    for (int k0 = 0; k0 < K; k0 += 32) {
        frag16 a0, a1, b[4];
        // --- A fragments: lane holds k = {k0+8h .. +7} and {k0+16+8h .. +7}
        a0.h[0] = *(const v8bf*)(pa0 + k0);
        a0.h[1] = *(const v8bf*)(pa0 + k0 + 16);
        a1.h[0] = *(const v8bf*)(pa1 + k0);
        a1.h[1] = *(const v8bf*)(pa1 + k0 + 16);
        __builtin_prefetch(pa0 + k0 + 32, 0, 1);   // global_prefetch_b8
        // --- B fragments: lane n holds W[n, k0+8h .. +7] and W[n, k0+16+8h .. +7]
#pragma unroll
        for (int j = 0; j < 4; ++j) {
            const __bf16* pb = W + (size_t)(n0 + j * 16 + l16) * K + k0 + half * 8;
            b[j].h[0] = *(const v8bf*)(pb);
            b[j].h[1] = *(const v8bf*)(pb + 16);
        }
#pragma unroll
        for (int j = 0; j < 4; ++j) {
            acc[0][j] = __builtin_amdgcn_wmma_f32_16x16x32_bf16(
                false, a0.v, false, b[j].v, (short)0, acc[0][j], false, false);
            acc[1][j] = __builtin_amdgcn_wmma_f32_16x16x32_bf16(
                false, a1.v, false, b[j].v, (short)0, acc[1][j], false, false);
        }
    }

    // Epilogue: C layout -> lane l16 = column, VGPR r -> row (r + 8*half)
#pragma unroll
    for (int i = 0; i < 2; ++i) {
#pragma unroll
        for (int j = 0; j < 4; ++j) {
            int n = n0 + j * 16 + l16;
            float bv = bias ? bias[n] : 0.0f;
#pragma unroll
            for (int r = 0; r < 8; ++r) {
                int m = m0 + wave * 32 + i * 16 + r + half * 8;
                if (m < M) {
                    float v = acc[i][j][r] + bv;
                    if (addend) v += addend[(size_t)m * N + n];
                    if (relu)   v = fmaxf(v, 0.0f);
                    C[(size_t)m * N + n] = v;
                }
            }
        }
    }
}

// ---------------------------------------------------------------------------
// Final classifier: out[l] = dot(h_mi[src[l]], h_di[dst[l]]) over 64 dims.
// One wave32 per label; lane k handles dims k and k+32; butterfly reduce.
// ---------------------------------------------------------------------------
__global__ void dot64_kernel(const float* __restrict__ hmi,
                             const float* __restrict__ hdi,
                             const int* __restrict__ ls,
                             const int* __restrict__ ld,
                             float* __restrict__ out, int L) {
    int wid = blockIdx.x * (blockDim.x >> 5) + (threadIdx.x >> 5);
    int lane = threadIdx.x & 31;
    if (wid >= L) return;
    const float* a = hmi + (size_t)ls[wid] * 64;
    const float* b = hdi + (size_t)ld[wid] * 64;
    float v = a[lane] * b[lane] + a[lane + 32] * b[lane + 32];
#pragma unroll
    for (int off = 16; off > 0; off >>= 1) v += __shfl_xor(v, off, 32);
    if (lane == 0) out[wid] = v;
}

// ---------------------------------------------------------------------------
// Launch
// ---------------------------------------------------------------------------
extern "C" void kernel_launch(void* const* d_in, const int* in_sizes, int n_in,
                              void* d_out, int out_size, void* d_ws, size_t ws_size,
                              hipStream_t stream) {
    const float* x_mi   = (const float*)d_in[0];
    const float* x_di   = (const float*)d_in[1];
    const float* W_mi   = (const float*)d_in[2];
    const float* b_mi   = (const float*)d_in[3];
    const float* W_di   = (const float*)d_in[4];
    const float* b_di   = (const float*)d_in[5];
    const float* emb_mi = (const float*)d_in[6];
    const float* emb_di = (const float*)d_in[7];

    const float* Wl[3][2]; const float* bl[3][2]; const float* Wr[3][2];
    int p = 8;
    for (int i = 0; i < 3; ++i)
        for (int r = 0; r < 2; ++r) {   // r==0: md, r==1: dm
            Wl[i][r] = (const float*)d_in[p++];
            bl[i][r] = (const float*)d_in[p++];
            Wr[i][r] = (const float*)d_in[p++];
        }
    const int* edge_src  = (const int*)d_in[26];
    const int* edge_dst  = (const int*)d_in[27];
    const int* label_src = (const int*)d_in[28];
    const int* label_dst = (const int*)d_in[29];
    float* out = (float*)d_out;

    // ---- Workspace carve-out ----
    float* ws = (float*)d_ws;
    size_t off = 0;
    auto carve = [&](size_t n) { float* q = ws + off; off += n; return q; };
    auto carve_bf = [&](size_t nelem) {          // bf16 region, 16B-aligned
        off = (off + 3) & ~(size_t)3;
        __bf16* q = (__bf16*)(ws + off);
        off += (nelem + 1) / 2;
        return q;
    };

    float* hA_mi = carve((size_t)N_MI * 256);
    float* hA_di = carve((size_t)N_DI * 256);
    float* hB_mi = carve((size_t)N_MI * 256);
    float* hB_di = carve((size_t)N_DI * 256);
    float* agg   = carve((size_t)N_MI * 256);
    float* deg_mi = carve(N_MI);
    float* deg_di = carve(N_DI);
    __bf16* bfA  = carve_bf((size_t)N_MI * 256);  // shared bf16 A operand

    auto zero = [&](float* ptr, long long n) {
        zero_kernel<<<dim3((unsigned)((n + 255) / 256)), 256, 0, stream>>>(ptr, n);
    };
    auto cvt = [&](const float* src, __bf16* dst, long long nelem) {
        long long n4 = nelem / 4;
        cvt_bf16_kernel<<<dim3((unsigned)((n4 + 255) / 256)), 256, 0, stream>>>(src, dst, n4);
    };
    // Pre-convert a weight matrix once per launch
    auto cvtw = [&](const float* w, size_t nelem) {
        __bf16* q = carve_bf(nelem);
        cvt(w, q, (long long)nelem);
        return q;
    };

    __bf16* bW_mi = cvtw(W_mi, (size_t)128 * 256);
    __bf16* bW_di = cvtw(W_di, (size_t)128 * 128);
    const int dims_in[3]  = {128, 256, 128};
    const int dims_out[3] = {256, 128, 64};
    __bf16* bWl[3][2]; __bf16* bWr[3][2];
    for (int i = 0; i < 3; ++i)
        for (int r = 0; r < 2; ++r) {
            size_t n = (size_t)dims_out[i] * dims_in[i];
            bWl[i][r] = cvtw(Wl[i][r], n);
            bWr[i][r] = cvtw(Wr[i][r], n);
        }

    // gemm: converts A to bf16 scratch, then WMMA GEMM (A fp32 in, C fp32 out)
    auto gemm = [&](const float* A32, const __bf16* Wb, const float* bias,
                    const float* add, float* C, int M, int N, int K, int relu) {
        cvt(A32, bfA, (long long)M * K);
        dim3 g((M + 127) / 128, N / 64);
        gemm_wmma_bf16<<<g, 128, 0, stream>>>(bfA, Wb, bias, add, C, M, N, K, relu);
    };

    // Degrees (edge structure only — constant across layers)
    zero(deg_mi, N_MI);
    zero(deg_di, N_DI);
    degree_kernel<<<(E_EDGES + 255) / 256, 256, 0, stream>>>(edge_src, edge_dst,
                                                             deg_mi, deg_di, E_EDGES);

    // Input projections: h = x @ W^T + b + emb
    gemm(x_mi, bW_mi, b_mi, emb_mi, hA_mi, N_MI, 128, 256, 0);
    gemm(x_di, bW_di, b_di, emb_di, hA_di, N_DI, 128, 128, 0);

    int Fin = 128;
    float* cur_mi = hA_mi; float* cur_di = hA_di;
    float* nxt_mi = hB_mi; float* nxt_di = hB_di;

    for (int li = 0; li < 3; ++li) {
        int Fout = dims_out[li];
        int relu = (li < 2) ? 1 : 0;

        // ---- disease update (md): aggregate h_mi over src->dst ----
        zero(agg, (long long)N_DI * Fin);
        scatter_kernel<<<8192, Fin, 0, stream>>>(cur_mi, edge_src, edge_dst, agg, E_EDGES, Fin);
        mean_kernel<<<(unsigned)(((long long)N_DI * Fin + 255) / 256), 256, 0, stream>>>(
            agg, deg_di, (long long)N_DI * Fin, Fin);
        gemm(agg,    bWl[li][0], bl[li][0], nullptr, nxt_di, N_DI, Fout, Fin, 0);
        gemm(cur_di, bWr[li][0], nullptr,   nxt_di,  nxt_di, N_DI, Fout, Fin, relu);

        // ---- miRNA update (dm): aggregate h_di over dst->src ----
        zero(agg, (long long)N_MI * Fin);
        scatter_kernel<<<8192, Fin, 0, stream>>>(cur_di, edge_dst, edge_src, agg, E_EDGES, Fin);
        mean_kernel<<<(unsigned)(((long long)N_MI * Fin + 255) / 256), 256, 0, stream>>>(
            agg, deg_mi, (long long)N_MI * Fin, Fin);
        gemm(agg,    bWl[li][1], bl[li][1], nullptr, nxt_mi, N_MI, Fout, Fin, 0);
        gemm(cur_mi, bWr[li][1], nullptr,   nxt_mi,  nxt_mi, N_MI, Fout, Fin, relu);

        // swap buffers
        float* t;
        t = cur_mi; cur_mi = nxt_mi; nxt_mi = t;
        t = cur_di; cur_di = nxt_di; nxt_di = t;
        Fin = Fout;
    }

    // Final dot product on label edges (64 dims) — one wave per label
    dot64_kernel<<<(L_LABELS + 7) / 8, 256, 0, stream>>>(cur_mi, cur_di,
                                                         label_src, label_dst,
                                                         out, L_LABELS);
    (void)in_sizes; (void)n_in; (void)out_size; (void)ws_size;
}
